// RNN_75299366633929
// MI455X (gfx1250) — matrix operand order repaired and checked by reading, not compiled, not run on hardware
//
#include <hip/hip_runtime.h>
#include <hip/hip_bf16.h>

// Problem constants
#define T_ 128
#define B_ 64
#define L_ 2
#define H_ 512
#define E_ 512
#define V_ 10000

typedef __bf16 bf16_t;
typedef bf16_t v16bf __attribute__((ext_vector_type(16)));
typedef float  v8f   __attribute__((ext_vector_type(8)));
typedef unsigned short ushort8  __attribute__((ext_vector_type(8)));
typedef unsigned short ushort16 __attribute__((ext_vector_type(16)));

static __device__ __forceinline__ unsigned short f2bf(float f) {
  unsigned int u = __builtin_bit_cast(unsigned int, f);
  u += 0x7FFFu + ((u >> 16) & 1u);   // round-to-nearest-even
  return (unsigned short)(u >> 16);
}

static __device__ __forceinline__ v8f zero8() {
  v8f z = {0.f, 0.f, 0.f, 0.f, 0.f, 0.f, 0.f, 0.f};
  return z;
}

// A-matrix 16x32 bf16 fragment (ISA 7.12.2): lane m=lane&15, hi=lane>>4,
// VGPR0-3 hold K = 8*hi..8*hi+7, VGPR4-7 hold K = 16+8*hi..23+8*hi.
static __device__ __forceinline__ v16bf fragA(const unsigned short* base, int row,
                                              int stride, int k0, int hi) {
  const unsigned short* p = base + row * stride + k0 + hi * 8;
  ushort8 lo  = *(const ushort8*)(p);
  ushort8 hi8 = *(const ushort8*)(p + 16);
  ushort16 c = __builtin_shufflevector(lo, hi8, 0, 1, 2, 3, 4, 5, 6, 7,
                                       8, 9, 10, 11, 12, 13, 14, 15);
  return __builtin_bit_cast(v16bf, c);
}

// B-matrix 32x16 bf16 fragment: lane n=lane&15, hi=lane>>4 holds K=8*hi..8*hi+7, 16+8*hi..23+8*hi
// (same order as A). W is stored (N,K) row-major, i.e. B supplied as N x K rows.
static __device__ __forceinline__ v16bf fragB(const unsigned short* base, int row,
                                              int stride, int k0, int hi) {
  const unsigned short* p = base + row * stride + k0 + hi * 8;
  ushort8 lo  = *(const ushort8*)(p);
  ushort8 hi8 = *(const ushort8*)(p + 16);
  ushort16 c = __builtin_shufflevector(lo, hi8, 0, 1, 2, 3, 4, 5, 6, 7,
                                       8, 9, 10, 11, 12, 13, 14, 15);
  return __builtin_bit_cast(v16bf, c);
}

static __device__ __forceinline__ v8f wmma_bf16(v16bf a, v16bf b, v8f c) {
  return __builtin_amdgcn_wmma_f32_16x16x32_bf16(false, a, false, b, (short)0, c,
                                                 false, false);
}

// ---------------- prep kernels ----------------

// emb_bf[(t*B+b)*E + e] = bf16(enc_w[inputs[t*B+b]*E + e])
__global__ void k_embed(const int* __restrict__ inputs, const float* __restrict__ enc_w,
                        unsigned short* __restrict__ emb_bf, int n) {
  for (int g = blockIdx.x * blockDim.x + threadIdx.x; g < n;
       g += gridDim.x * blockDim.x) {
    int tb = g >> 9;          // / E_
    int e  = g & (E_ - 1);
    emb_bf[g] = f2bf(enc_w[inputs[tb] * E_ + e]);
  }
}

// wcat[n][k] (512 x 1024): k<512 -> reg_w0[n][k], else rec_w[0][n][k-512]
__global__ void k_wcat(const float* __restrict__ reg_w0, const float* __restrict__ rec_w,
                       unsigned short* __restrict__ wcat) {
  const int n_elems = H_ * (E_ + H_);
  for (int g = blockIdx.x * blockDim.x + threadIdx.x; g < n_elems;
       g += gridDim.x * blockDim.x) {
    int nrow = g >> 10;
    int k    = g & 1023;
    float v = (k < E_) ? reg_w0[nrow * E_ + k] : rec_w[nrow * H_ + (k - E_)];
    wcat[g] = f2bf(v);
  }
}

// wsum = reg_w[0] + rec_w[1]  (512 x 512)
__global__ void k_wsum(const float* __restrict__ reg_w, const float* __restrict__ rec_w,
                       unsigned short* __restrict__ wsum) {
  const int n_elems = H_ * H_;
  for (int g = blockIdx.x * blockDim.x + threadIdx.x; g < n_elems;
       g += gridDim.x * blockDim.x) {
    wsum[g] = f2bf(reg_w[g] + rec_w[H_ * H_ + g]);
  }
}

__global__ void k_decw(const float* __restrict__ dec_w,
                       unsigned short* __restrict__ decw_bf, int n) {
  for (int g = blockIdx.x * blockDim.x + threadIdx.x; g < n;
       g += gridDim.x * blockDim.x) {
    decw_bf[g] = f2bf(dec_w[g]);
  }
}

// ---------------- recurrence kernel ----------------
// 8 blocks, one per 16-row block of the (L*B)=128 stacked rows; 8 waves per
// block split the N=512 columns (4 N-tiles of 16 each). All T steps in one
// launch; activations live in LDS as bf16; weights stream from L2 every step
// (kept loop-variant via an asm memory clobber so LICM cannot spill them).
__global__ void __launch_bounds__(256) k_recur(
    const unsigned short* __restrict__ emb,    // (T,B,E) bf16
    const unsigned short* __restrict__ wcat,   // (H, E+H) bf16
    const unsigned short* __restrict__ wsum,   // (H, H) bf16
    const float* __restrict__ reg_b0,          // (H)
    const float* __restrict__ rec_b,           // (L,H)
    const float* __restrict__ hidden_init,     // (L,B,H) f32
    unsigned short* __restrict__ xdec,         // (T,B,H) bf16 out
    float* __restrict__ h_final)               // (L,B,H) f32 out
{
  __shared__ __align__(16) unsigned short sh_h[16 * H_];
  __shared__ __align__(16) unsigned short sh_x[16 * H_];

  const int rb  = blockIdx.x;      // 0..7 ; rows 16*rb .. 16*rb+15 of (L*B)
  const int tid = threadIdx.x;

  for (int i = tid; i < 16 * H_; i += 256) {
    int mr = i >> 9, k = i & (H_ - 1);
    sh_h[i] = f2bf(hidden_init[(rb * 16 + mr) * H_ + k]);
  }
  __syncthreads();

  const int w    = tid >> 5;
  const int lane = tid & 31;
  const int m    = lane & 15;
  const int hi   = lane >> 4;
  const int bm   = (rb & 3) * 16 + m;   // batch index of this lane's A row
  const int ntB  = w * 4;               // this wave's first N-tile (of 32)
  const bool top = (rb >= 4);           // l == L-1 rows feed the decoder

  for (int t = 0; t < T_; ++t) {
    // Stop LICM from hoisting+spilling the t-invariant weight fragments:
    // keep streaming them from L2 each step instead.
    asm volatile("" ::: "memory");

    const unsigned short* embT = emb + (size_t)(t * B_) * E_;

    // ---- layer 0: tanh(emb @ reg_w0^T + h @ rec_w0^T + reg_b0 + rec_b[0]) ----
    v8f acc0 = zero8(), acc1 = zero8(), acc2 = zero8(), acc3 = zero8();
    for (int kk = 0; kk < 32; ++kk) {   // K = E + H = 1024, step 32
      v16bf a = (kk < 16) ? fragA(embT, bm, E_, kk * 32, hi)
                          : fragA(sh_h, m, H_, (kk - 16) * 32, hi);
      v16bf b0 = fragB(wcat, (ntB + 0) * 16 + m, E_ + H_, kk * 32, hi);
      v16bf b1 = fragB(wcat, (ntB + 1) * 16 + m, E_ + H_, kk * 32, hi);
      v16bf b2 = fragB(wcat, (ntB + 2) * 16 + m, E_ + H_, kk * 32, hi);
      v16bf b3 = fragB(wcat, (ntB + 3) * 16 + m, E_ + H_, kk * 32, hi);
      acc0 = wmma_bf16(a, b0, acc0);
      acc1 = wmma_bf16(a, b1, acc1);
      acc2 = wmma_bf16(a, b2, acc2);
      acc3 = wmma_bf16(a, b3, acc3);
    }
    {
      v8f accs[4] = {acc0, acc1, acc2, acc3};
#pragma unroll
      for (int j = 0; j < 4; ++j) {
        int n = (ntB + j) * 16 + m;
        float bias = reg_b0[n] + rec_b[n];
#pragma unroll
        for (int v = 0; v < 8; ++v) {
          float x = tanhf(accs[j][v] + bias);
          sh_x[(v + 8 * hi) * H_ + n] = f2bf(x);
        }
      }
    }
    __syncthreads();   // sh_x complete; sh_h fully consumed

    // ---- layer 1: tanh(x0 @ (reg_w[0]+rec_w[1])^T + rec_b[1]) ----
    acc0 = zero8(); acc1 = zero8(); acc2 = zero8(); acc3 = zero8();
    for (int kk = 0; kk < 16; ++kk) {   // K = H = 512
      v16bf a  = fragA(sh_x, m, H_, kk * 32, hi);
      v16bf b0 = fragB(wsum, (ntB + 0) * 16 + m, H_, kk * 32, hi);
      v16bf b1 = fragB(wsum, (ntB + 1) * 16 + m, H_, kk * 32, hi);
      v16bf b2 = fragB(wsum, (ntB + 2) * 16 + m, H_, kk * 32, hi);
      v16bf b3 = fragB(wsum, (ntB + 3) * 16 + m, H_, kk * 32, hi);
      acc0 = wmma_bf16(a, b0, acc0);
      acc1 = wmma_bf16(a, b1, acc1);
      acc2 = wmma_bf16(a, b2, acc2);
      acc3 = wmma_bf16(a, b3, acc3);
    }
    {
      v8f accs[4] = {acc0, acc1, acc2, acc3};
#pragma unroll
      for (int j = 0; j < 4; ++j) {
        int n = (ntB + j) * 16 + m;
        float bias = rec_b[H_ + n];
#pragma unroll
        for (int v = 0; v < 8; ++v) {
          float x = tanhf(accs[j][v] + bias);
          int mr = v + 8 * hi;
          sh_h[mr * H_ + n] = f2bf(x);                     // next-step hidden
          if (top)
            xdec[(size_t)(t * B_ + (rb - 4) * 16 + mr) * H_ + n] = f2bf(x);
          if (t == T_ - 1)
            h_final[(rb * 16 + mr) * H_ + n] = x;
        }
      }
    }
    __syncthreads();   // sh_h (new) complete before next step reads it
  }
}

// ---------------- decoder GEMM ----------------
// out(8192 x 10000) = xdec(8192 x 512 bf16) @ decw(10000 x 512 bf16)^T + dec_b.
// WG = 256 threads = 8 waves laid out 2(M) x 4(N); WG tile 128 x 256;
// each wave computes a 64 x 64 tile (16 accumulators). Per K-step: issue all
// 8 fragment loads first, then 16 back-to-back WMMAs (one wait per K-step).
__global__ void __launch_bounds__(256) k_decode(
    const unsigned short* __restrict__ xdec,
    const unsigned short* __restrict__ decw,
    const float* __restrict__ dec_b,
    float* __restrict__ out)
{
  const int tid  = threadIdx.x;
  const int w    = tid >> 5;
  const int lane = tid & 31;
  const int m    = lane & 15;
  const int hi   = lane >> 4;
  const int M0   = blockIdx.x * 128 + (w >> 2) * 64;
  const int N0   = blockIdx.y * 256 + (w & 3) * 64;

  v8f acc[4][4];
#pragma unroll
  for (int i = 0; i < 4; ++i)
#pragma unroll
    for (int j = 0; j < 4; ++j) acc[i][j] = zero8();

  for (int kk = 0; kk < 16; ++kk) {    // K = 512, step 32
    // Prefetch next K-panel of activations (shared across the 4 N-waves).
    if (kk < 15)
      __builtin_prefetch((const void*)(xdec + (size_t)(M0 + m) * H_ + (kk + 1) * 32),
                         0, 1);

    v16bf afr[4];
#pragma unroll
    for (int i = 0; i < 4; ++i)
      afr[i] = fragA(xdec, M0 + i * 16 + m, H_, kk * 32, hi);

    v16bf bfr[4];
#pragma unroll
    for (int j = 0; j < 4; ++j) {
      int n  = N0 + j * 16 + m;
      int nc = (n < V_) ? n : (V_ - 1);      // clamp pad rows (never stored)
      bfr[j] = fragB(decw, nc, H_, kk * 32, hi);
    }

#pragma unroll
    for (int i = 0; i < 4; ++i)
#pragma unroll
      for (int j = 0; j < 4; ++j) acc[i][j] = wmma_bf16(afr[i], bfr[j], acc[i][j]);
  }

#pragma unroll
  for (int i = 0; i < 4; ++i) {
#pragma unroll
    for (int j = 0; j < 4; ++j) {
      int n = N0 + j * 16 + m;
      if (n < V_) {
        float bias = dec_b[n];
#pragma unroll
        for (int v = 0; v < 8; ++v) {
          int row = M0 + i * 16 + v + 8 * hi;
          out[(size_t)row * V_ + n] = acc[i][j][v] + bias;
        }
      }
    }
  }
}

// ---------------- launch ----------------
extern "C" void kernel_launch(void* const* d_in, const int* in_sizes, int n_in,
                              void* d_out, int out_size, void* d_ws, size_t ws_size,
                              hipStream_t stream) {
  const int*   inputs = (const int*)d_in[0];
  const float* hidden = (const float*)d_in[1];
  const float* enc_w  = (const float*)d_in[2];
  const float* dec_w  = (const float*)d_in[3];
  const float* dec_b  = (const float*)d_in[4];
  const float* reg_w0 = (const float*)d_in[5];
  const float* reg_b0 = (const float*)d_in[6];
  const float* reg_w  = (const float*)d_in[7];
  const float* rec_w  = (const float*)d_in[8];
  const float* rec_b  = (const float*)d_in[9];
  float* out = (float*)d_out;

  char* ws = (char*)d_ws;
  unsigned short* emb_bf  = (unsigned short*)(ws + 0);          //  8,388,608 B
  unsigned short* xdec_bf = (unsigned short*)(ws + 8388608);    //  8,388,608 B
  unsigned short* wcat_bf = (unsigned short*)(ws + 16777216);   //  1,048,576 B
  unsigned short* wsum_bf = (unsigned short*)(ws + 17825792);   //    524,288 B
  unsigned short* decw_bf = (unsigned short*)(ws + 18350080);   // 10,240,000 B

  k_embed<<<2048, 256, 0, stream>>>(inputs, enc_w, emb_bf, T_ * B_ * E_);
  k_wcat <<<1024, 256, 0, stream>>>(reg_w0, rec_w, wcat_bf);
  k_wsum <<<512,  256, 0, stream>>>(reg_w, rec_w, wsum_bf);
  k_decw <<<4096, 256, 0, stream>>>(dec_w, decw_bf, V_ * H_);

  float* h_final = out + (size_t)T_ * B_ * V_;
  k_recur<<<8, 256, 0, stream>>>(emb_bf, wcat_bf, wsum_bf, reg_b0, rec_b,
                                 hidden, xdec_bf, h_final);

  dim3 grid(8192 / 128, (V_ + 255) / 256);   // 64 x 40
  k_decode<<<grid, 256, 0, stream>>>(xdec_bf, decw_bf, dec_b, out);
}
